// PWCDCNet_90082644066609
// MI455X (gfx1250) — matrix operand order repaired and checked by reading, not compiled, not run on hardware
//
#include <hip/hip_runtime.h>
#include <hip/hip_bf16.h>
#include <stdint.h>
#include <stddef.h>

// ---------------------------------------------------------------------------
// CDNA5 / gfx1250 PWC-Net forward.  All convolutions (pyramid, dense flow
// blocks, prediction heads, context net) run through one implicit-GEMM kernel
// built on v_wmma_f32_16x16x32_f16 (wave32, 16x16 C tile per wave, f32 acc),
// with the A tile staged through LDS (ds_store_b128 / ds_load_b128) and the
// hot loop free of integer division (tap x channel-chunk iteration).
// ---------------------------------------------------------------------------

typedef __attribute__((ext_vector_type(16))) _Float16 v16h;
typedef __attribute__((ext_vector_type(8)))  float    v8f;
typedef __attribute__((ext_vector_type(4)))  float    f32x4;

#define LEAK 0.1f

__device__ __forceinline__ float lrelu_f(float x) { return x > 0.f ? x : LEAK * x; }
__device__ __forceinline__ int   iclamp(int v, int lo, int hi) { return v < lo ? lo : (v > hi ? hi : v); }

// ---------------------------------------------------------------------------
// Pack HWIO f32 weights -> f16 [Npad][Kpad], K = (ky*KW+kx)*Cpad + ci,
// Cpad = ceil32(Cin), zero fill for ci>=Cin / n>=Cout.  Row-major per output
// channel so each lane's 16 consecutive K values are 16B-aligned b128 loads.
// ---------------------------------------------------------------------------
__global__ void pack_weights_f16(const float* __restrict__ w, _Float16* __restrict__ out,
                                 int Cin, int Cpad, int Cout, int Kpad, int Npad)
{
    int t = blockIdx.x * blockDim.x + threadIdx.x;
    int total = Npad * Kpad;
    if (t >= total) return;
    int n = t / Kpad;
    int k = t - n * Kpad;
    float v = 0.f;
    if (n < Cout) {
        int ci = k % Cpad;
        int kk = k / Cpad;                       // ky*KW+kx
        if (ci < Cin) v = w[((size_t)kk * Cin + ci) * Cout + n];
    }
    out[(size_t)n * Kpad + k] = (_Float16)v;
}

// ---------------------------------------------------------------------------
// Implicit-GEMM conv, NHWC, SAME padding, stride/dilation, fused bias+lrelu.
// Reads a channel slice [cin_off, cin_off+Cin) of a tensor with Cin_t total
// channels; writes slice [cout_off, cout_off+Cout) of a Cout_t tensor
// (zero-copy DenseNet concatenation).
// Block = 256 threads = 8 waves; wave (mi,ni) owns 16x16 of the 64x32 C tile.
// Loop structure: taps (ky,kx incremental, no div) x 32-channel chunks.
// A tile (64 rows x 32 K, f16, 4KB) staged in LDS: one 8-channel run per
// thread cooperative load, then two ds_load_b128 per lane in WMMA layout.
// ---------------------------------------------------------------------------
__global__ void __launch_bounds__(256)
conv2d_wmma(const float* __restrict__ src, int Hin, int Win, int Cin_t, int cin_off,
            int Cin, int Cpad,
            const _Float16* __restrict__ wp, int Kpad,
            const float* __restrict__ bias,
            float* __restrict__ dst, int Hout, int Wout, int Cout_t, int cout_off, int Cout,
            int ntaps, int KW, int stride, int dil, int padH, int padW, int relu, int M)
{
    __shared__ _Float16 As[64 * 32];

    const int tid  = threadIdx.x;
    const int lane = tid & 31;
    const int wave = tid >> 5;
    const int part = lane >> 4;          // half-wave (selects K sub-range)
    const int l16  = lane & 15;

    const int m0 = blockIdx.y * 64 + (wave & 3) * 16;
    const int n0 = blockIdx.x * 32 + (wave >> 2) * 16;
    const int hw = Hout * Wout;

    // Cooperative A-loader identity: one 8-channel run per thread.
    const int arow = tid >> 2;           // 0..63  (A tile row)
    const int arun = tid & 3;            // 0..3   (K offset arun*8 in chunk)
    const int am   = blockIdx.y * 64 + arow;
    int ab = 0, aoy = 0, aox = 0;
    const bool avalid = (am < M);
    if (avalid) {
        ab = am / hw;
        int r = am - ab * hw;
        aoy = r / Wout;
        aox = r - aoy * Wout;
    }

    v8f acc = {0.f, 0.f, 0.f, 0.f, 0.f, 0.f, 0.f, 0.f};

    const _Float16* brow = wp + (size_t)(n0 + l16) * Kpad + part * 16;
    const _Float16* arp  = As + ((wave & 3) * 16 + l16) * 32 + part * 16;

    int ky = 0, kx = 0;
    for (int tap = 0; tap < ntaps; ++tap) {
        // Hoisted per-tap spatial position for the cooperative loader.
        const int iy = aoy * stride + ky * dil - padH;
        const int ix = aox * stride + kx * dil - padW;
        const bool ok = avalid && (unsigned)iy < (unsigned)Hin && (unsigned)ix < (unsigned)Win;
        const float* prow = ok
            ? src + ((size_t)(ab * Hin + iy) * Win + ix) * Cin_t + cin_off
            : src;
        const _Float16* btap = brow + (size_t)tap * Cpad;

        for (int ci0 = 0; ci0 < Cpad; ci0 += 32) {
            // ---- cooperative global->reg load of this thread's 8-run -------
            const int c = ci0 + arun * 8;
            float vals[8] = {0.f, 0.f, 0.f, 0.f, 0.f, 0.f, 0.f, 0.f};
            if (ok) {
                if (c + 8 <= Cin) {
#pragma unroll
                    for (int j = 0; j < 8; ++j) vals[j] = prow[c + j];
                } else {
#pragma unroll
                    for (int j = 0; j < 8; ++j) vals[j] = (c + j < Cin) ? prow[c + j] : 0.f;
                }
            }
            __syncthreads();             // previous chunk's LDS reads complete
#pragma unroll
            for (int j = 0; j < 8; ++j) As[arow * 32 + arun * 8 + j] = (_Float16)vals[j];
            __syncthreads();             // A tile visible to all waves

            // ---- A fragment from LDS (contiguous K order, same as B) ------
            // lanes 0-15 : elems 0-15 = K 0..15
            // lanes 16-31: elems 0-15 = K 16..31
            union { f32x4 q[2]; v16h h; } af;
            af.q[0] = *(const f32x4*)(arp);
            af.q[1] = *(const f32x4*)(arp + 8);

            // ---- B fragment: lane n holds 16 consecutive K (b128 x2) -------
            union { f32x4 q[2]; v16h h; } bf;
            const _Float16* bp = btap + ci0;
            bf.q[0] = *(const f32x4*)(bp);
            bf.q[1] = *(const f32x4*)(bp + 8);
            __builtin_prefetch(bp + 32, 0, 1);   // global_prefetch_b8: next chunk

            acc = __builtin_amdgcn_wmma_f32_16x16x32_f16(
                      false, af.h, false, bf.h, (short)0, acc, false, false);
        }
        if (++kx == KW) { kx = 0; ++ky; }
    }

    // ---- epilogue: bias + leaky relu, scatter 16x16 tile -------------------
    const int nn = n0 + l16;
    const float bv = (nn < Cout) ? bias[nn] : 0.f;
#pragma unroll
    for (int r = 0; r < 8; ++r) {
        int mm = m0 + part * 8 + r;              // lanes 0-15: M=r, 16-31: M=8+r
        if (mm < M && nn < Cout) {
            int bb = mm / hw;
            int rr = mm - bb * hw;
            int yy = rr / Wout;
            int xx = rr - yy * Wout;
            float v = acc[r] + bv;
            if (relu) v = lrelu_f(v);
            dst[((size_t)(bb * Hout + yy) * Wout + xx) * Cout_t + cout_off + nn] = v;
        }
    }
}

// ---------------------------------------------------------------------------
// Cost volume: 81 displacements, mean over channels, fused leaky relu,
// written directly into the concat buffer at channel offset coff.
// ---------------------------------------------------------------------------
__global__ void cost_volume_lrelu(const float* __restrict__ c1, const float* __restrict__ c2,
                                  int B, int H, int W, int C,
                                  float* __restrict__ dst, int Ct, int coff, int total)
{
    (void)B;
    int t = blockIdx.x * blockDim.x + threadIdx.x;
    if (t >= total) return;
    int d = t % 81;
    int p = t / 81;
    int x = p % W;  p /= W;
    int y = p % H;
    int b = p / H;
    int dy = d / 9 - 4, dx = d % 9 - 4;
    int y2 = y + dy, x2 = x + dx;
    float s = 0.f;
    if ((unsigned)y2 < (unsigned)H && (unsigned)x2 < (unsigned)W) {
        const float* a = c1 + ((size_t)(b * H + y) * W + x) * C;
        const float* e = c2 + ((size_t)(b * H + y2) * W + x2) * C;
        float acc = 0.f;
        for (int c = 0; c < C; ++c) acc += a[c] * e[c];
        s = acc / (float)C;
    }
    dst[((size_t)(b * H + y) * W + x) * Ct + coff + d] = lrelu_f(s);
}

// ---------------------------------------------------------------------------
// Bilinear dense warp; flow read from a channel slice of the concat buffer.
// ---------------------------------------------------------------------------
__global__ void warp_bilinear(const float* __restrict__ img, int B, int H, int W, int C,
                              const float* __restrict__ flow, int Ft, int foff, float scale,
                              float* __restrict__ dst, int total)
{
    (void)B;
    int t = blockIdx.x * blockDim.x + threadIdx.x;
    if (t >= total) return;
    int c = t % C;
    int p = t / C;
    int x = p % W;  p /= W;
    int y = p % H;
    int b = p / H;
    const float* fp = flow + ((size_t)(b * H + y) * W + x) * Ft + foff;
    float xs = (float)x + fp[0] * scale;
    float ys = (float)y + fp[1] * scale;
    float x0f = floorf(xs), y0f = floorf(ys);
    float wx = xs - x0f, wy = ys - y0f;
    int x0 = iclamp((int)x0f, 0, W - 1);
    int x1 = iclamp(x0 + 1, 0, W - 1);
    int y0 = iclamp((int)y0f, 0, H - 1);
    int y1 = iclamp(y0 + 1, 0, H - 1);
    const size_t rb = (size_t)b * H;
    float v00 = img[((rb + y0) * W + x0) * C + c];
    float v01 = img[((rb + y0) * W + x1) * C + c];
    float v10 = img[((rb + y1) * W + x0) * C + c];
    float v11 = img[((rb + y1) * W + x1) * C + c];
    dst[((size_t)(b * H + y) * W + x) * C + c] =
        v00 * (1.f - wx) * (1.f - wy) + v01 * wx * (1.f - wy) +
        v10 * (1.f - wx) * wy + v11 * wx * wy;
}

// ---------------------------------------------------------------------------
// 4x4 stride-2 conv-transpose, 2 output channels (upflow / upfeat heads).
// ---------------------------------------------------------------------------
__global__ void deconv4x4s2(const float* __restrict__ src, int Hin, int Win, int Cin,
                            const float* __restrict__ w, const float* __restrict__ bias,
                            float* __restrict__ dst, int Ct, int coff, int total)
{
    int t = blockIdx.x * blockDim.x + threadIdx.x;
    if (t >= total) return;
    int Ho = 2 * Hin, Wo = 2 * Win;
    int co = t & 1;
    int p = t >> 1;
    int ox = p % Wo;  p /= Wo;
    int oy = p % Ho;
    int b  = p / Ho;
    float s = bias[co];
    for (int ky = 0; ky < 4; ++ky) {
        int ty = oy + 1 - ky;
        if (ty < 0 || (ty & 1)) continue;
        int iy = ty >> 1;
        if (iy >= Hin) continue;
        for (int kx = 0; kx < 4; ++kx) {
            int tx = ox + 1 - kx;
            if (tx < 0 || (tx & 1)) continue;
            int ix = tx >> 1;
            if (ix >= Win) continue;
            const float* ip = src + ((size_t)(b * Hin + iy) * Win + ix) * Cin;
            const float* wpk = w + ((size_t)(ky * 4 + kx)) * Cin * 2 + co;
            for (int ci = 0; ci < Cin; ++ci) s += ip[ci] * wpk[(size_t)ci * 2];
        }
    }
    dst[((size_t)(b * Ho + oy) * Wo + ox) * Ct + coff + co] = s;
}

// ---------------------------------------------------------------------------
__global__ void slice_copy(const float* __restrict__ src, int Cs,
                           float* __restrict__ dst, int Ct, int coff, int total)
{
    int t = blockIdx.x * blockDim.x + threadIdx.x;
    if (t >= total) return;
    int c = t % Cs;
    int p = t / Cs;
    dst[(size_t)p * Ct + coff + c] = src[t];
}

__global__ void add_inplace(float* __restrict__ a, const float* __restrict__ b, int n)
{
    int t = blockIdx.x * blockDim.x + threadIdx.x;
    if (t < n) a[t] += b[t];
}

// ---------------------------------------------------------------------------
// 4x bilinear upsample (half-pixel centers) * 20.0 -> final flow output.
// ---------------------------------------------------------------------------
__global__ void upsample4x_x20(const float* __restrict__ src, int B, int H, int W,
                               float* __restrict__ dst, int total)
{
    (void)B;
    int t = blockIdx.x * blockDim.x + threadIdx.x;
    if (t >= total) return;
    int Ho = 4 * H, Wo = 4 * W;
    int c = t & 1;
    int p = t >> 1;
    int ox = p % Wo;  p /= Wo;
    int oy = p % Ho;
    int b  = p / Ho;
    float sx = ((float)ox + 0.5f) * 0.25f - 0.5f;
    float sy = ((float)oy + 0.5f) * 0.25f - 0.5f;
    float x0f = floorf(sx), y0f = floorf(sy);
    float wx = sx - x0f, wy = sy - y0f;
    int x0 = iclamp((int)x0f, 0, W - 1);
    int x1 = iclamp(x0 + 1, 0, W - 1);
    int y0 = iclamp((int)y0f, 0, H - 1);
    int y1 = iclamp(y0 + 1, 0, H - 1);
    const size_t rb = (size_t)b * H;
    float v00 = src[((rb + y0) * W + x0) * 2 + c];
    float v01 = src[((rb + y0) * W + x1) * 2 + c];
    float v10 = src[((rb + y1) * W + x0) * 2 + c];
    float v11 = src[((rb + y1) * W + x1) * 2 + c];
    float v = v00 * (1.f - wx) * (1.f - wy) + v01 * wx * (1.f - wy) +
              v10 * (1.f - wx) * wy + v11 * wx * wy;
    dst[((size_t)(b * Ho + oy) * Wo + ox) * 2 + c] = v * 20.f;
}

// ---------------------------------------------------------------------------
// Host orchestration
// ---------------------------------------------------------------------------
namespace {
struct Bump {
    char*  base;
    size_t off;
    float*    f(size_t n) { size_t o = off; off = (off + n * 4 + 255) & ~(size_t)255; return (float*)(base + o); }
    _Float16* h(size_t n) { size_t o = off; off = (off + n * 2 + 255) & ~(size_t)255; return (_Float16*)(base + o); }
};
struct FB { int base, pb, pw, ufb, ufw, uxb, uxw; };
} // namespace

extern "C" void kernel_launch(void* const* d_in, const int* in_sizes, int n_in,
                              void* d_out, int out_size, void* d_ws, size_t ws_size,
                              hipStream_t stream)
{
    (void)in_sizes; (void)n_in; (void)out_size; (void)ws_size;
    auto P = [&](int i) { return (const float*)d_in[i]; };
    const float* img1 = P(0);
    const float* img2 = P(1);

    const int B = 4;
    const int Hl[7] = {384, 192, 96, 48, 24, 12, 6};
    const int Wl[7] = {512, 256, 128, 64, 32, 16, 8};
    const int Cl[7] = {3, 16, 32, 64, 96, 128, 192};

    // Flattened pytree index map (top-level insertion order; dict keys sorted):
    //   image_1, image_2, then params: ctx, fb2, fb3, fb4, fb5, fb6, pyr.
    //   ctx: convs[(w,b)x6]=2..13, pb=14, pw=15
    //   fbN: dense[(w,b)x5], pb, pw, (ufb, ufw, uxb, uxw)
    const FB fbs[5] = {
        {76, 86, 87, 88, 89, 90, 91},   // fb6
        {60, 70, 71, 72, 73, 74, 75},   // fb5
        {44, 54, 55, 56, 57, 58, 59},   // fb4
        {28, 38, 39, 40, 41, 42, 43},   // fb3
        {16, 26, 27, -1, -1, -1, -1},   // fb2 (output block)
    };
    const int PYR = 92;                 // level l: b0,b1,b2,w0,w1,w2 at 92+6*(l-1)

    Bump ws{(char*)d_ws, 0};
    _Float16* wscr = ws.h((size_t)1 << 20);   // 2 MB weight-packing scratch (stream-serialized reuse)

    auto conv = [&](const float* src, int Hin, int Win, int Cin_t, int cin_off, int Cin,
                    const float* w, const float* bias,
                    float* dst, int Hout, int Wout, int Cout_t, int cout_off, int Cout,
                    int KH, int KW, int stride, int dil, int relu) {
        int Cpad = (Cin + 31) & ~31;              // chunk-aligned channels
        int Kpad = KH * KW * Cpad;
        int Npad = (Cout + 31) & ~31;
        int tot  = Npad * Kpad;
        pack_weights_f16<<<dim3((tot + 255) / 256), dim3(256), 0, stream>>>(
            w, wscr, Cin, Cpad, Cout, Kpad, Npad);
        int M   = B * Hout * Wout;
        int ptH = (Hout - 1) * stride + (KH - 1) * dil + 1 - Hin; if (ptH < 0) ptH = 0;
        int ptW = (Wout - 1) * stride + (KW - 1) * dil + 1 - Win; if (ptW < 0) ptW = 0;
        dim3 g(Npad / 32, (M + 63) / 64);
        conv2d_wmma<<<g, dim3(256), 0, stream>>>(
            src, Hin, Win, Cin_t, cin_off, Cin, Cpad, wscr, Kpad, bias,
            dst, Hout, Wout, Cout_t, cout_off, Cout,
            KH * KW, KW, stride, dil, ptH / 2, ptW / 2, relu, M);
    };

    // ---------------- feature pyramids --------------------------------------
    float* feat[2][7];
    float* tA = ws.f((size_t)B * 192 * 256 * 16);
    float* tB = ws.f((size_t)B * 192 * 256 * 16);
    for (int im = 0; im < 2; ++im) {
        const float* x = im ? img2 : img1;
        int pc = 3;
        for (int l = 1; l <= 6; ++l) {
            int H = Hl[l], W = Wl[l], C = Cl[l];
            feat[im][l] = ws.f((size_t)B * H * W * C);
            int bs = PYR + 6 * (l - 1);
            conv(x,  Hl[l - 1], Wl[l - 1], pc, 0, pc, P(bs + 3), P(bs + 0), tA, H, W, C, 0, C, 3, 3, 2, 1, 1);
            conv(tA, H, W, C, 0, C,                 P(bs + 4), P(bs + 1), tB, H, W, C, 0, C, 3, 3, 1, 1, 1);
            conv(tB, H, W, C, 0, C,                 P(bs + 5), P(bs + 2), feat[im][l], H, W, C, 0, C, 3, 3, 1, 1, 1);
            x = feat[im][l];
            pc = C;
        }
    }

    // Dense flow block: prepend-concat in one buffer, channel offsets
    // [o5(32)|o4(64)|o3(96)|o2(128)|o1(128)|x0(in_ch)], x0 at 448.
    const int FBD[5] = {128, 128, 96, 64, 32};
    auto flow_dense = [&](float* X, int H, int W, int Ctot, int in_ch, int base) {
        int cin = in_ch, coff = 448;
        for (int j = 0; j < 5; ++j) {
            int f = FBD[j], noff = coff - f;
            conv(X, H, W, Ctot, coff, cin, P(base + 2 * j), P(base + 2 * j + 1),
                 X, H, W, Ctot, noff, f, 3, 3, 1, 1, 1);
            cin += f;
            coff = noff;
        }
    };

    // ---------------- level 6 -----------------------------------------------
    int H6 = Hl[6], W6 = Wl[6];
    int in6 = 81, Ct6 = in6 + 448;
    float* X6 = ws.f((size_t)B * H6 * W6 * Ct6);
    {
        int tot = B * H6 * W6 * 81;
        cost_volume_lrelu<<<dim3((tot + 255) / 256), dim3(256), 0, stream>>>(
            feat[0][6], feat[1][6], B, H6, W6, Cl[6], X6, Ct6, 448, tot);
    }
    flow_dense(X6, H6, W6, Ct6, in6, fbs[0].base);
    float* flow6 = ws.f((size_t)B * H6 * W6 * 2);
    conv(X6, H6, W6, Ct6, 0, Ct6, P(fbs[0].pw), P(fbs[0].pb), flow6, H6, W6, 2, 0, 2, 3, 3, 1, 1, 0);

    // ---------------- levels 5..2 -------------------------------------------
    const float scales[5] = {0.f, 0.625f, 1.25f, 2.5f, 5.0f};
    float* Xc = X6; int Ctc = Ct6; float* flowc = flow6; FB fbc = fbs[0];
    float* flow2 = nullptr;
    for (int i = 1; i < 5; ++i) {
        int l = 6 - i;
        int H = Hl[l], W = Wl[l], C = Cl[l];
        int in = 81 + C + 4, Ct = in + 448;
        float* X = ws.f((size_t)B * H * W * Ct);
        int offU = 448 + 81 + C;       // upflow channels; upfeat at offU+2
        {
            int tot = B * H * W * 2;
            deconv4x4s2<<<dim3((tot + 255) / 256), dim3(256), 0, stream>>>(
                flowc, Hl[l + 1], Wl[l + 1], 2, P(fbc.ufw), P(fbc.ufb), X, Ct, offU, tot);
            deconv4x4s2<<<dim3((tot + 255) / 256), dim3(256), 0, stream>>>(
                Xc, Hl[l + 1], Wl[l + 1], Ctc, P(fbc.uxw), P(fbc.uxb), X, Ct, offU + 2, tot);
        }
        float* wrp = ws.f((size_t)B * H * W * C);
        {
            int tot = B * H * W * C;
            warp_bilinear<<<dim3((tot + 255) / 256), dim3(256), 0, stream>>>(
                feat[1][l], B, H, W, C, X, Ct, offU, scales[i], wrp, tot);
        }
        {
            int tot = B * H * W * 81;
            cost_volume_lrelu<<<dim3((tot + 255) / 256), dim3(256), 0, stream>>>(
                feat[0][l], wrp, B, H, W, C, X, Ct, 448, tot);
        }
        {
            int tot = B * H * W * C;
            slice_copy<<<dim3((tot + 255) / 256), dim3(256), 0, stream>>>(
                feat[0][l], C, X, Ct, 448 + 81, tot);
        }
        FB f = fbs[i];
        flow_dense(X, H, W, Ct, in, f.base);
        float* fl = ws.f((size_t)B * H * W * 2);
        conv(X, H, W, Ct, 0, Ct, P(f.pw), P(f.pb), fl, H, W, 2, 0, 2, 3, 3, 1, 1, 0);
        Xc = X; Ctc = Ct; flowc = fl; fbc = f;
        if (l == 2) flow2 = fl;
    }

    // ---------------- context network ---------------------------------------
    int H2 = Hl[2], W2 = Wl[2];
    float* cA = ws.f((size_t)B * H2 * W2 * 128);
    float* cB = ws.f((size_t)B * H2 * W2 * 128);
    const int dils[6] = {1, 2, 4, 8, 16, 1};
    const int cch[7]  = {2, 128, 128, 128, 96, 64, 32};
    const float* cx = flow2;
    float* bufs[2] = {cA, cB};
    for (int j = 0; j < 6; ++j) {
        conv(cx, H2, W2, cch[j], 0, cch[j], P(2 + 2 * j), P(3 + 2 * j),
             bufs[j & 1], H2, W2, cch[j + 1], 0, cch[j + 1], 3, 3, 1, dils[j], 1);
        cx = bufs[j & 1];
    }
    float* cpred = ws.f((size_t)B * H2 * W2 * 2);
    conv(cx, H2, W2, 32, 0, 32, P(15), P(14), cpred, H2, W2, 2, 0, 2, 3, 3, 1, 1, 0);
    {
        int n = B * H2 * W2 * 2;
        add_inplace<<<dim3((n + 255) / 256), dim3(256), 0, stream>>>(flow2, cpred, n);
    }
    {
        int tot = B * (4 * H2) * (4 * W2) * 2;
        upsample4x_x20<<<dim3((tot + 255) / 256), dim3(256), 0, stream>>>(
            flow2, B, H2, W2, (float*)d_out, tot);
    }
}
